// HistogramLoss_65231963292288
// MI455X (gfx1250) — matrix operand order repaired and checked by reading, not compiled, not run on hardware
//
#include <hip/hip_runtime.h>

typedef __attribute__((ext_vector_type(16))) _Float16 v16h;
typedef __attribute__((ext_vector_type(8)))  float    v8f;

#define N_ROWS 4096
#define D_DIM  512
#define NSTEPS 151
#define HALFB  75
#define INV_STEP 75.0f        /* 1/STEP, STEP = 2/150 */

#define BM 128
#define BN 128
#define BK 32
#define LDA 40                /* f16 LDS row stride (32 data + 8 pad) */
#define HPAD (NSTEPS + 1)     /* 152 */

union frag_t { unsigned u[8]; v16h v; };

// ---------------------------------------------------------------------------
// Kernel 0: zero global histograms
// ---------------------------------------------------------------------------
__global__ void zero_kernel(float* hp, float* hn) {
    int t = blockIdx.x * blockDim.x + threadIdx.x;
    if (t < NSTEPS) { hp[t] = 0.0f; hn[t] = 0.0f; }
}

// ---------------------------------------------------------------------------
// Kernel 1: L2-normalize rows, emit f16.  One wave32 per row, 8 rows/block.
// ---------------------------------------------------------------------------
__global__ __launch_bounds__(256) void norm_kernel(const float* __restrict__ feats,
                                                   _Float16* __restrict__ out) {
    const int wave = threadIdx.x >> 5;
    const int lane = threadIdx.x & 31;
    const int row  = blockIdx.x * 8 + wave;

    const float4* src = (const float4*)(feats + (size_t)row * D_DIM);
    float4 v[4];
    float ss = 0.0f;
#pragma unroll
    for (int c = 0; c < 4; ++c) {
        v[c] = src[lane + 32 * c];
        ss += v[c].x * v[c].x + v[c].y * v[c].y + v[c].z * v[c].z + v[c].w * v[c].w;
    }
#pragma unroll
    for (int off = 16; off >= 1; off >>= 1) ss += __shfl_xor(ss, off, 32);
    const float r = rsqrtf(ss);

    _Float16* dst = out + (size_t)row * D_DIM;
#pragma unroll
    for (int c = 0; c < 4; ++c) {
        union { _Float16 h[4]; uint2 u; } pk;
        pk.h[0] = (_Float16)(v[c].x * r);
        pk.h[1] = (_Float16)(v[c].y * r);
        pk.h[2] = (_Float16)(v[c].z * r);
        pk.h[3] = (_Float16)(v[c].w * r);
        *(uint2*)(dst + (lane + 32 * c) * 4) = pk.u;
    }
}

// ---------------------------------------------------------------------------
// Kernel 2: double-buffered f16 WMMA GEMM over upper-tri block pairs + hist
// ---------------------------------------------------------------------------
__global__ __launch_bounds__(256) void gemm_hist_kernel(const _Float16* __restrict__ F,
                                                        const int* __restrict__ cls,
                                                        float* __restrict__ hp,
                                                        float* __restrict__ hn) {
    __shared__ _Float16 As[2][BM * LDA];
    __shared__ _Float16 Bs[2][BN * LDA];
    __shared__ float    hpos[8][HPAD];
    __shared__ float    hneg[8][HPAD];
    __shared__ int      ci[BM];
    __shared__ int      cj[BN];

    // decode linear block id -> (bi, bj), bi <= bj, over 32x32 block grid
    const int nb = N_ROWS / BM;            // 32
    int p = blockIdx.x;
    int bi = 0;
    while (p >= nb - bi) { p -= nb - bi; ++bi; }
    const int bj    = bi + p;
    const int ibase = bi * BM;
    const int jbase = bj * BN;

    const int tid  = threadIdx.x;
    const int wave = tid >> 5;
    const int lane = tid & 31;
    const int lg   = lane >> 4;            // lane group 0/1
    const int m    = lane & 15;

    // zero per-wave histograms, load class labels
    for (int t = tid; t < 8 * HPAD; t += 256) {
        ((float*)hpos)[t] = 0.0f;
        ((float*)hneg)[t] = 0.0f;
    }
    if (tid < BM) { ci[tid] = cls[ibase + tid]; cj[tid] = cls[jbase + tid]; }

    v8f acc[8];
#pragma unroll
    for (int t = 0; t < 8; ++t) acc[t] = {};

    const int ldr = tid >> 1;              // LDS tile row this thread fills
    const int ldh = tid & 1;               // which 16-f16 half of the 32-wide row
    const _Float16* gA = F + (size_t)(ibase + ldr) * D_DIM + ldh * 16;
    const _Float16* gB = F + (size_t)(jbase + ldr) * D_DIM + ldh * 16;
    const int ldsOff = ldr * LDA + ldh * 16;

    // prologue: stage panel 0 into buffer 0 (also covers hist-zero/cls loads)
    {
        const uint4* ga = (const uint4*)gA;
        const uint4* gb = (const uint4*)gB;
        uint4 a0 = ga[0], a1 = ga[1], b0 = gb[0], b1 = gb[1];
        uint4* sa = (uint4*)(As[0] + ldsOff);
        uint4* sb = (uint4*)(Bs[0] + ldsOff);
        sa[0] = a0; sa[1] = a1; sb[0] = b0; sb[1] = b1;
    }
    __syncthreads();

    int buf = 0;
    for (int kc = 0; kc < D_DIM; kc += BK, buf ^= 1) {
        const bool has_next = (kc + BK < D_DIM);
        uint4 a0, a1, b0, b1;
        if (has_next) {                    // next panel in flight during WMMAs
            const uint4* ga = (const uint4*)(gA + kc + BK);
            const uint4* gb = (const uint4*)(gB + kc + BK);
            a0 = ga[0]; a1 = ga[1]; b0 = gb[0]; b1 = gb[1];
            if (kc + 2 * BK < D_DIM) {     // warm L2/L0 for the panel after
                __builtin_prefetch(gA + kc + 2 * BK, 0, 3);
                __builtin_prefetch(gB + kc + 2 * BK, 0, 3);
            }
        }

        // A fragment: lane l (g=l>>4, M=l&15); VGPR v: K=(v<4 ? 2v : 16+2(v-4))+8g
        frag_t af;
        const unsigned* Arow = (const unsigned*)(As[buf] + (wave * 16 + m) * LDA);
#pragma unroll
        for (int vv = 0; vv < 8; ++vv) {
            const int k0 = (vv < 4 ? 2 * vv : 16 + 2 * (vv - 4)) + 8 * lg;
            af.u[vv] = Arow[k0 >> 1];
        }

        // 8 column tiles in two batches of 4: loads pipeline ahead of WMMAs
#pragma unroll
        for (int half = 0; half < 2; ++half) {
            frag_t bf[4];
#pragma unroll
            for (int t = 0; t < 4; ++t) {
                const unsigned* Brow =
                    (const unsigned*)(Bs[buf] + ((half * 4 + t) * 16 + m) * LDA);
#pragma unroll
                for (int vv = 0; vv < 8; ++vv) bf[t].u[vv] = Brow[vv + (vv & 4) + 4 * lg];
            }
#pragma unroll
            for (int t = 0; t < 4; ++t)
                acc[half * 4 + t] = __builtin_amdgcn_wmma_f32_16x16x32_f16(
                    false, af.v, false, bf[t].v, (short)0, acc[half * 4 + t],
                    false, false);
        }

        if (has_next) {                    // commit next panel to other buffer
            uint4* sa = (uint4*)(As[buf ^ 1] + ldsOff);
            uint4* sb = (uint4*)(Bs[buf ^ 1] + ldsOff);
            sa[0] = a0; sa[1] = a1; sb[0] = b0; sb[1] = b1;
        }
        __syncthreads();
    }

    // soft-binning: C layout VGPR e, lane l -> M = e + 8*(l>=16), N = l&15
#pragma unroll
    for (int t = 0; t < 8; ++t) {
        const int jl  = t * 16 + m;
        const int gj  = jbase + jl;
        const int cjv = cj[jl];
#pragma unroll
        for (int e = 0; e < 8; ++e) {
            const int il = wave * 16 + e + 8 * lg;
            const int gi = ibase + il;
            if (gi < gj) {
                const float s    = acc[t][e];
                const float x    = s * INV_STEP;
                const float kf   = floorf(x);
                const float frac = x - kf;
                int ik = (int)kf + HALFB;
                int lo = ik < 0 ? 0 : (ik > NSTEPS - 1 ? NSTEPS - 1 : ik);
                int hi = ik + 1 < 0 ? 0 : (ik + 1 > NSTEPS - 1 ? NSTEPS - 1 : ik + 1);
                float* h = (ci[il] == cjv) ? hpos[wave] : hneg[wave];
                atomicAdd(&h[lo], 1.0f - frac);
                atomicAdd(&h[hi], frac);
            }
        }
    }
    __syncthreads();

    // reduce per-wave LDS hists -> global
    if (tid < NSTEPS) {
        float sp = 0.0f, sn = 0.0f;
#pragma unroll
        for (int w = 0; w < 8; ++w) { sp += hpos[w][tid]; sn += hneg[w][tid]; }
        atomicAdd(&hp[tid], sp);
        atomicAdd(&hn[tid], sn);
    }
}

// ---------------------------------------------------------------------------
// Kernel 3: normalize hists (sums == mask counts), cumsum, dot -> scalar loss
// ---------------------------------------------------------------------------
__global__ void finalize_kernel(const float* __restrict__ hp,
                                const float* __restrict__ hn,
                                float* __restrict__ out) {
    if (threadIdx.x == 0 && blockIdx.x == 0) {
        float sp = 0.0f, sn = 0.0f;
        for (int j = 0; j < NSTEPS; ++j) { sp += hp[j]; sn += hn[j]; }
        const float isp = 1.0f / sp, isn = 1.0f / sn;
        float cum = 0.0f, loss = 0.0f;
        for (int j = 0; j < NSTEPS; ++j) {
            cum  += hp[j] * isp;
            loss += hn[j] * isn * cum;
        }
        out[0] = loss;
    }
}

// ---------------------------------------------------------------------------
extern "C" void kernel_launch(void* const* d_in, const int* in_sizes, int n_in,
                              void* d_out, int out_size, void* d_ws, size_t ws_size,
                              hipStream_t stream) {
    const float* feats   = (const float*)d_in[0];
    const int*   classes = (const int*)d_in[1];

    _Float16* f16 = (_Float16*)d_ws;
    float* hp = (float*)((char*)d_ws + (size_t)N_ROWS * D_DIM * sizeof(_Float16));
    float* hn = hp + 256;

    zero_kernel<<<1, 256, 0, stream>>>(hp, hn);
    norm_kernel<<<N_ROWS / 8, 256, 0, stream>>>(feats, f16);

    const int nb      = N_ROWS / BM;          // 32
    const int nblocks = nb * (nb + 1) / 2;    // 528 upper-tri block pairs
    gemm_hist_kernel<<<nblocks, 256, 0, stream>>>(f16, classes, hp, hn);

    finalize_kernel<<<1, 32, 0, stream>>>(hp, hn, (float*)d_out);
}
